// Actor_54408645705760
// MI455X (gfx1250) — matrix operand order repaired and checked by reading, not compiled, not run on hardware
//
#include <hip/hip_runtime.h>
#include <math.h>

// ---------------------------------------------------------------------------
// Types for CDNA5 WMMA (wave32): D(16x16 f32) = A(16x32 bf16) x B(32x16 bf16)
// ---------------------------------------------------------------------------
typedef __bf16 v16bf __attribute__((ext_vector_type(16)));
typedef float  f32x8 __attribute__((ext_vector_type(8)));

union Frag {
    uint4 q[2];   // two 16B chunks (loaded with b128)
    v16bf v;      // 32 bytes = 16 bf16 = 8 VGPRs
};

__device__ __forceinline__ unsigned short f2bf(float f) {
    // round-to-nearest-even fp32 -> bf16
    unsigned int u = __float_as_uint(f);
    u += 0x7FFFu + ((u >> 16) & 1u);
    return (unsigned short)(u >> 16);
}

__device__ __forceinline__ f32x8 wmma_bf16(const Frag& a, const Frag& b, f32x8 c) {
    return __builtin_amdgcn_wmma_f32_16x16x32_bf16(
        /*neg_a=*/false, a.v, /*neg_b=*/false, b.v,
        /*c_mod=*/(short)0, c, /*reuse_a=*/false, /*reuse_b=*/false);
}

// ---------------------------------------------------------------------------
// Problem constants
// ---------------------------------------------------------------------------
#define BATCH      65536
#define SDIM       1024
#define HDIM       2048
#define ODIM       256       // 2*ACTION_DIM
#define ADIM       128
#define BM         32        // batch rows per workgroup
#define SX         1032      // LDS stride (bf16 elems) for x tile   (1024+8)
#define SH         2056      // LDS stride (bf16 elems) for h tiles  (2048+8)
#define SO         264       // LDS stride (f32 elems) for out stage (256+8)

// ---------------------------------------------------------------------------
// BN statistics, stage 1: per-block partial sums / sumsq over 256-row chunks
// grid=256 blocks x 256 threads; partials at part[b*2048 + col] (+1024 sumsq)
// ---------------------------------------------------------------------------
__global__ __launch_bounds__(256) void bn_partial(const float* __restrict__ states,
                                                  float* __restrict__ part) {
    const int t = threadIdx.x;
    const int b = blockIdx.x;
    const int col = t * 4;
    float4 s = make_float4(0.f, 0.f, 0.f, 0.f);
    float4 q = make_float4(0.f, 0.f, 0.f, 0.f);
    const float* p = states + (size_t)b * 256 * SDIM + col;
    for (int r = 0; r < 256; ++r) {
        float4 v = *(const float4*)(p + (size_t)r * SDIM);
        s.x += v.x; s.y += v.y; s.z += v.z; s.w += v.w;
        q.x += v.x * v.x; q.y += v.y * v.y; q.z += v.z * v.z; q.w += v.w * v.w;
    }
    float* o = part + (size_t)b * 2048;
    *(float4*)(o + col)        = s;
    *(float4*)(o + 1024 + col) = q;
}

// ---------------------------------------------------------------------------
// BN statistics, stage 2: reduce partials, build scale/shift
// grid=4 x 256 -> 1024 columns
// ---------------------------------------------------------------------------
__global__ __launch_bounds__(256) void bn_finalize(const float* __restrict__ part,
                                                   const float* __restrict__ bnw,
                                                   const float* __restrict__ bnb,
                                                   float* __restrict__ scale,
                                                   float* __restrict__ shift) {
    const int c = blockIdx.x * 256 + threadIdx.x;
    float s = 0.f, q = 0.f;
    for (int b = 0; b < 256; ++b) {
        s += part[(size_t)b * 2048 + c];
        q += part[(size_t)b * 2048 + 1024 + c];
    }
    const float inv = 1.0f / (float)BATCH;
    const float mean = s * inv;
    const float var  = q * inv - mean * mean;
    const float sc   = bnw[c] * rsqrtf(var + 1e-5f);
    scale[c] = sc;
    shift[c] = bnb[c] - mean * sc;
}

// ---------------------------------------------------------------------------
// fp32 -> bf16 weight conversion + swizzle into WMMA B-fragment blocks.
// For a layer [N x K] (row-major, D = x @ W^T): one 512-element (1KB) block
// per (n_tile, k_step), block index b = n_tile*(K/32) + k_step. Within a
// block, lane l owns elements l*16..l*16+15, which are
//   W[n_tile*16 + (l&15)][k_step*32 + (e>>3)*16 + (l>>4)*8 + (e&7)],  e = 0..15
// i.e. exactly the V_WMMA_*_BF16 B-operand register file for that lane.
// A B-fragment load is then 2x global_load_b128 at (block*1KB + lane*32B).
// ---------------------------------------------------------------------------
__global__ __launch_bounds__(256) void wswizzle(const float* __restrict__ w1,
                                                const float* __restrict__ w2,
                                                const float* __restrict__ w3,
                                                unsigned short* __restrict__ o) {
    const size_t N1 = (size_t)HDIM * SDIM;      // 2,097,152
    const size_t N2 = (size_t)HDIM * HDIM;      // 4,194,304
    const size_t N3 = (size_t)ODIM * HDIM;      //   524,288
    size_t i = (size_t)blockIdx.x * 256 + threadIdx.x;
    if (i >= N1 + N2 + N3) return;
    const float* src;
    size_t p;
    int K;
    if (i < N1)           { src = w1; p = i;            K = SDIM; }
    else if (i < N1 + N2) { src = w2; p = i - N1;       K = HDIM; }
    else                  { src = w3; p = i - N1 - N2;  K = HDIM; }
    const int sk     = K >> 5;            // K/32 steps
    const size_t blk = p >> 9;            // /512 elements per block
    const int within = (int)(p & 511);
    const int lane   = within >> 4;
    const int e      = within & 15;
    const int nt     = (int)(blk / sk);
    const int ks     = (int)(blk % sk);
    const int row    = nt * 16 + (lane & 15);
    const int col    = ks * 32 + (e >> 3) * 16 + (lane >> 4) * 8 + (e & 7);
    o[i] = f2bf(src[(size_t)row * K + col]);
}

// ---------------------------------------------------------------------------
// One hidden layer: out(32 x 2048) = relu(A(32 x K) @ W^T + bias), bf16 WMMA.
// A in LDS (row-major bf16, stride aStride), W pre-swizzled B-fragment blocks
// in global. Wave w owns N columns [w*256, w*256+256) in 4 register blocks of
// (2 M-tiles x 4 N-tiles) = 64 accumulator VGPRs. SK = K/32.
// ---------------------------------------------------------------------------
template <int SK>
__device__ __forceinline__ void layer_gemm(const unsigned short* aLds, const int aStride,
                                           const unsigned short* __restrict__ wB,
                                           const float* __restrict__ bias,
                                           unsigned short* oLds, const int oStride,
                                           const int wave, const int lane) {
    const int lrow = lane & 15;
    const int lhi  = (lane >> 4) & 1;
#pragma unroll 1
    for (int nb = 0; nb < 4; ++nb) {
        const int ntBase = wave * 16 + nb * 4;
        // base of this wave's fragment stream for n-tile ntBase, k-step 0
        const unsigned short* bp0 = wB + (size_t)ntBase * SK * 512 + lane * 16;
        f32x8 acc[2][4] = {};
#pragma unroll 2
        for (int ks = 0; ks < SK; ++ks) {
            Frag a[2];
#pragma unroll
            for (int mt = 0; mt < 2; ++mt) {
                const int base = (mt * 16 + lrow) * aStride + ks * 32 + lhi * 8;
                a[mt].q[0] = *(const uint4*)(aLds + base);
                a[mt].q[1] = *(const uint4*)(aLds + base + 16);
            }
            Frag b[4];
#pragma unroll
            for (int nt = 0; nt < 4; ++nt) {
                // nt offset and ks offset are compile-time-scaled immediates
                const unsigned short* bp = bp0 + ((size_t)nt * SK + ks) * 512;
                b[nt].q[0] = *(const uint4*)(bp);
                b[nt].q[1] = *(const uint4*)(bp + 8);
            }
#pragma unroll
            for (int mt = 0; mt < 2; ++mt)
#pragma unroll
                for (int nt = 0; nt < 4; ++nt)
                    acc[mt][nt] = wmma_bf16(a[mt], b[nt], acc[mt][nt]);
        }
        // epilogue: bias + relu, bf16 back to LDS (row-major, A-ready for next layer)
#pragma unroll
        for (int mt = 0; mt < 2; ++mt)
#pragma unroll
            for (int nt = 0; nt < 4; ++nt) {
                const int col = (ntBase + nt) * 16 + lrow;
                const float bv = bias[col];
#pragma unroll
                for (int v = 0; v < 8; ++v) {
                    float r = acc[mt][nt][v] + bv;
                    r = fmaxf(r, 0.0f);
                    oLds[(mt * 16 + v + lhi * 8) * oStride + col] = f2bf(r);
                }
            }
    }
}

// ---------------------------------------------------------------------------
// Fully fused actor forward: BN-apply -> L1 -> L2 -> L3 -> tanh -> atan2/pi
// 2048 blocks x 256 threads (8 waves). LDS: ldsA (x, then h2), ldsB (h1, then
// f32 tanh staging). 2 x 128.5KB = 257KB of the 320KB WGP LDS.
// ---------------------------------------------------------------------------
__global__ __launch_bounds__(256) void actor_fused(
    const float* __restrict__ states,
    const float* __restrict__ scale, const float* __restrict__ shift,
    const unsigned short* __restrict__ w1b, const float* __restrict__ b1,
    const unsigned short* __restrict__ w2b, const float* __restrict__ b2,
    const unsigned short* __restrict__ w3b, const float* __restrict__ b3,
    float* __restrict__ out) {
    __shared__ __align__(16) unsigned short ldsA[BM * SH];
    __shared__ __align__(16) unsigned short ldsB[BM * SH];

    const int tid  = threadIdx.x;
    const int lane = tid & 31;
    const int wave = tid >> 5;
    const int row0 = blockIdx.x * BM;

    // ---- stage normalized x tile (bf16, row-major, stride SX) into ldsA ----
    {
        const int col = tid * 4;  // 256 threads * 4 = 1024 cols
        const float4 sc = *(const float4*)(scale + col);
        const float4 sh = *(const float4*)(shift + col);
#pragma unroll 4
        for (int r = 0; r < BM; ++r) {
            float4 v = *(const float4*)(states + (size_t)(row0 + r) * SDIM + col);
            unsigned int p0 = (unsigned int)f2bf(v.x * sc.x + sh.x) |
                              ((unsigned int)f2bf(v.y * sc.y + sh.y) << 16);
            unsigned int p1 = (unsigned int)f2bf(v.z * sc.z + sh.z) |
                              ((unsigned int)f2bf(v.w * sc.w + sh.w) << 16);
            uint2 p; p.x = p0; p.y = p1;
            *(uint2*)(ldsA + r * SX + col) = p;
        }
    }
    __syncthreads();

    // ---- layer 1: h1 = relu(x @ w1^T + b1)   (K=1024) : ldsA -> ldsB ----
    layer_gemm<32>(ldsA, SX, w1b, b1, ldsB, SH, wave, lane);
    __syncthreads();

    // ---- layer 2: h2 = relu(h1 @ w2^T + b2)  (K=2048) : ldsB -> ldsA ----
    layer_gemm<64>(ldsB, SH, w2b, b2, ldsA, SH, wave, lane);
    __syncthreads();

    // ---- layer 3: o = tanh(h2 @ w3^T + b3)   (N=256)  : ldsA -> f32 ldsB ----
    float* oF = reinterpret_cast<float*>(ldsB);
    {
        const int lrow = lane & 15;
        const int lhi  = (lane >> 4) & 1;
        const int ntBase = wave * 2;  // 8 waves x 2 tiles x 16 = 256 cols
        const unsigned short* bp0 = w3b + (size_t)ntBase * 64 * 512 + lane * 16;
        f32x8 acc[2][2] = {};
#pragma unroll 2
        for (int ks = 0; ks < 64; ++ks) {
            Frag a[2];
#pragma unroll
            for (int mt = 0; mt < 2; ++mt) {
                const int base = (mt * 16 + lrow) * SH + ks * 32 + lhi * 8;
                a[mt].q[0] = *(const uint4*)(ldsA + base);
                a[mt].q[1] = *(const uint4*)(ldsA + base + 16);
            }
            Frag b[2];
#pragma unroll
            for (int nt = 0; nt < 2; ++nt) {
                const unsigned short* bp = bp0 + ((size_t)nt * 64 + ks) * 512;
                b[nt].q[0] = *(const uint4*)(bp);
                b[nt].q[1] = *(const uint4*)(bp + 8);
            }
#pragma unroll
            for (int mt = 0; mt < 2; ++mt)
#pragma unroll
                for (int nt = 0; nt < 2; ++nt)
                    acc[mt][nt] = wmma_bf16(a[mt], b[nt], acc[mt][nt]);
        }
#pragma unroll
        for (int mt = 0; mt < 2; ++mt)
#pragma unroll
            for (int nt = 0; nt < 2; ++nt) {
                const int col = (ntBase + nt) * 16 + lrow;
                const float bv = b3[col];
#pragma unroll
                for (int v = 0; v < 8; ++v)
                    oF[(mt * 16 + v + lhi * 8) * SO + col] = tanhf(acc[mt][nt][v] + bv);
            }
    }
    __syncthreads();

    // ---- epilogue: out[m, j] = atan2(o[m,2j], o[m,2j+1]) / pi ----
#pragma unroll
    for (int j = 0; j < 16; ++j) {
        const int idx = j * 256 + tid;       // 32*128 = 4096 results
        const int r = idx >> 7;
        const int c = idx & 127;
        const float y  = oF[r * SO + 2 * c];
        const float xv = oF[r * SO + 2 * c + 1];
        out[(size_t)(row0 + r) * ADIM + c] = atan2f(y, xv) * 0.3183098861837907f;
    }
}

// ---------------------------------------------------------------------------
// Host-side launcher (graph-capture safe: only kernel launches on `stream`)
// ---------------------------------------------------------------------------
extern "C" void kernel_launch(void* const* d_in, const int* in_sizes, int n_in,
                              void* d_out, int out_size, void* d_ws, size_t ws_size,
                              hipStream_t stream) {
    const float* states = (const float*)d_in[0];
    const float* bnw    = (const float*)d_in[1];
    const float* bnb    = (const float*)d_in[2];
    const float* w1     = (const float*)d_in[3];
    const float* b1     = (const float*)d_in[4];
    const float* w2     = (const float*)d_in[5];
    const float* b2     = (const float*)d_in[6];
    const float* w3     = (const float*)d_in[7];
    const float* b3     = (const float*)d_in[8];
    float* out          = (float*)d_out;

    // workspace layout (floats): [0,524288) partials | scale 1024 | shift 1024 |
    // pad | swizzled bf16 weights (w1|w2|w3) = 6,815,744 ushorts (~13 MB).
    float* ws    = (float*)d_ws;
    float* part  = ws;
    float* scale = ws + 524288;
    float* shift = scale + 1024;
    unsigned short* w1b = (unsigned short*)(ws + 526336);  // 16B-aligned offset
    unsigned short* w2b = w1b + (size_t)HDIM * SDIM;
    unsigned short* w3b = w2b + (size_t)HDIM * HDIM;

    bn_partial <<<256, 256, 0, stream>>>(states, part);
    bn_finalize<<<4, 256, 0, stream>>>(part, bnw, bnb, scale, shift);
    wswizzle   <<<26624, 256, 0, stream>>>(w1, w2, w3, w1b);
    actor_fused<<<BATCH / BM, 256, 0, stream>>>(states, scale, shift,
                                                w1b, b1, w2b, b2, w3b, b3, out);
}
